// MambaViT_41850161332554
// MI455X (gfx1250) — matrix operand order repaired and checked by reading, not compiled, not run on hardware
//
#include <hip/hip_runtime.h>
#include <hip/hip_bf16.h>
#include <math.h>

// ---------------------------------------------------------------------------
// MambaViT forward for MI455X (gfx1250, wave32, WMMA).
// GEMMs: fp32 -> fp16 LDS tiles, v_wmma_f32_16x16x32_f16, f32 accumulate.
//   128x64 block tile, 4 waves, each wave a 32x64 strip = 8 WMMA / K-step,
//   next-tile global_prefetch_b8 issued before the barrier.
// Scan: one lane per (batch, channel), 16-wide SSM state in VGPRs.
// ---------------------------------------------------------------------------

#define BB     16
#define CC     3
#define HH     224
#define WW     224
#define PP     16
#define DIM    256
#define NL     8
#define DS     16
#define DCONV  4
#define NCLS   1000
#define DI     512          // 2*DIM
#define DTR    16
#define NPATCH 196          // 14*14
#define SEQ    197
#define MROWS  (BB*SEQ)     // 3152
#define PD     768          // C*P*P

typedef _Float16 h16;
typedef __attribute__((ext_vector_type(16))) _Float16 v16h;
typedef __attribute__((ext_vector_type(8)))  float    v8f;

// ------------------------- WMMA tiled GEMM ---------------------------------
// C[M,N] = A[M,K] @ B[K,N] (+bias). K must be a multiple of 32.
// Block: 128 threads = 4 waves; 128x64 output tile; each wave owns a 32x64
// strip -> 2 A fragments, 4 B fragments, 8 v_wmma per 32-deep K step.
#define TM 128
#define TN 64
#define TK 32

__global__ __launch_bounds__(128)
void k_gemm_wmma(const float* __restrict__ A, int lda,
                 const float* __restrict__ Bw, int ldb,
                 float* __restrict__ C, int ldc,
                 int M, int N, int K,
                 const float* __restrict__ bias)
{
    __shared__ h16 As[TM][TK + 8];   // row-major A tile
    __shared__ h16 Bs[TN][TK + 8];   // B tile stored transposed: Bs[n][k]

    const int tid  = threadIdx.x;
    const int lane = tid & 31;
    const int wave = tid >> 5;
    const int bm   = blockIdx.y * TM;
    const int bn   = blockIdx.x * TN;

    v8f acc[2][4];
#pragma unroll
    for (int f = 0; f < 2; ++f)
#pragma unroll
        for (int j = 0; j < 4; ++j) acc[f][j] = {};

    // tile-load split: A 128x32 -> 32 elems/thread; B 32x64 -> 16 elems/thread
    const int ar  = tid;                 // A tile row 0..127
    const int bk  = tid >> 2;            // B tile k row 0..31
    const int bn0 = (tid & 3) * 16;      // B tile col base 0/16/32/48

    for (int kt = 0; kt < K; kt += TK) {
        // --- stage A (128x32, f32 -> f16) ---
        {
            const int gr = bm + ar;
            if (gr < M) {
                const float* src = A + (size_t)gr * lda + kt;
#pragma unroll
                for (int i = 0; i < 32; ++i) As[ar][i] = (h16)src[i];
                if (kt + TK < K) __builtin_prefetch(src + TK, 0, 0);
            } else {
#pragma unroll
                for (int i = 0; i < 32; ++i) As[ar][i] = (h16)0.f;
            }
        }
        // --- stage B (32x64) transposed into Bs[n][k] ---
        {
            const float* src = Bw + (size_t)(kt + bk) * ldb + bn + bn0;
#pragma unroll
            for (int i = 0; i < 16; ++i) {
                const int gn = bn + bn0 + i;
                Bs[bn0 + i][bk] = (gn < N) ? (h16)src[i] : (h16)0.f;
            }
            if (kt + TK < K) __builtin_prefetch(src + (size_t)TK * ldb, 0, 0);
        }
        __syncthreads();

        // --- A fragments: 16x32, M = lane&15, K pairs per ISA layout ---
        const int lm = lane & 15;
        const int kb = (lane >> 4) * 8;
        const int r0 = wave * 32 + lm;
        v16h a0, a1;
#pragma unroll
        for (int i = 0; i < 4; ++i) {
            a0[2*i]         = As[r0][kb + 2*i];
            a0[2*i + 1]     = As[r0][kb + 2*i + 1];
            a0[8 + 2*i]     = As[r0][kb + 16 + 2*i];
            a0[8 + 2*i + 1] = As[r0][kb + 16 + 2*i + 1];
            a1[2*i]         = As[r0 + 16][kb + 2*i];
            a1[2*i + 1]     = As[r0 + 16][kb + 2*i + 1];
            a1[8 + 2*i]     = As[r0 + 16][kb + 16 + 2*i];
            a1[8 + 2*i + 1] = As[r0 + 16][kb + 16 + 2*i + 1];
        }

        // --- B fragments: 32x16 each, N = lane&15, K = kb + i + (i & 8) (same as A) ---
        const int kb2 = (lane >> 4) * 16;
        v16h b0, b1, b2, b3;
#pragma unroll
        for (int i = 0; i < 16; ++i) b0[i] = Bs[ 0 + lm][kb + i + (i & 8)];
#pragma unroll
        for (int i = 0; i < 16; ++i) b1[i] = Bs[16 + lm][kb + i + (i & 8)];
#pragma unroll
        for (int i = 0; i < 16; ++i) b2[i] = Bs[32 + lm][kb + i + (i & 8)];
#pragma unroll
        for (int i = 0; i < 16; ++i) b3[i] = Bs[48 + lm][kb + i + (i & 8)];

        acc[0][0] = __builtin_amdgcn_wmma_f32_16x16x32_f16(false, a0, false, b0,
                                                           (short)0, acc[0][0], false, false);
        acc[1][0] = __builtin_amdgcn_wmma_f32_16x16x32_f16(false, a1, false, b0,
                                                           (short)0, acc[1][0], false, false);
        acc[0][1] = __builtin_amdgcn_wmma_f32_16x16x32_f16(false, a0, false, b1,
                                                           (short)0, acc[0][1], false, false);
        acc[1][1] = __builtin_amdgcn_wmma_f32_16x16x32_f16(false, a1, false, b1,
                                                           (short)0, acc[1][1], false, false);
        acc[0][2] = __builtin_amdgcn_wmma_f32_16x16x32_f16(false, a0, false, b2,
                                                           (short)0, acc[0][2], false, false);
        acc[1][2] = __builtin_amdgcn_wmma_f32_16x16x32_f16(false, a1, false, b2,
                                                           (short)0, acc[1][2], false, false);
        acc[0][3] = __builtin_amdgcn_wmma_f32_16x16x32_f16(false, a0, false, b3,
                                                           (short)0, acc[0][3], false, false);
        acc[1][3] = __builtin_amdgcn_wmma_f32_16x16x32_f16(false, a1, false, b3,
                                                           (short)0, acc[1][3], false, false);
        __syncthreads();
    }

    // --- epilogue: C/D layout -> M = 8*(lane>=16)+vgpr, N = lane&15 ---
    const int lm = lane & 15;
    const int mb0 = bm + wave * 32 + (lane >> 4) * 8;
#pragma unroll
    for (int f = 0; f < 2; ++f) {
#pragma unroll
        for (int j = 0; j < 4; ++j) {
            const int n = bn + j * 16 + lm;
            if (n < N) {
                const float bv = bias ? bias[n] : 0.f;
#pragma unroll
                for (int r = 0; r < 8; ++r) {
                    const int m = mb0 + f * 16 + r;
                    if (m < M) C[(size_t)m * ldc + n] = acc[f][j][r] + bv;
                }
            }
        }
    }
}

// -------------------- patch extraction + LN over 768 -----------------------
__global__ __launch_bounds__(256)
void k_patch_ln(const float* __restrict__ img, const float* __restrict__ g,
                const float* __restrict__ bt, float* __restrict__ out)
{
    const int p  = blockIdx.x;            // 0..3135
    const int b  = p / NPATCH;
    const int pj = p % NPATCH;
    const int ti = pj / 14, tj = pj % 14;
    const int tid = threadIdx.x;
    __shared__ float red[256];

    float v[3]; float s = 0.f;
#pragma unroll
    for (int e = 0; e < 3; ++e) {
        const int f  = tid * 3 + e;                 // 0..767
        const int c  = f % 3;
        const int pp = f / 3;
        const int px = pp % 16, py = pp / 16;
        const int row = ti * 16 + py, col = tj * 16 + px;
        v[e] = img[(((size_t)b * CC + c) * HH + row) * WW + col];
        s += v[e];
    }
    red[tid] = s; __syncthreads();
    for (int o = 128; o > 0; o >>= 1) { if (tid < o) red[tid] += red[tid + o]; __syncthreads(); }
    const float mu = red[0] * (1.f / PD);
    __syncthreads();
    float vs = 0.f;
#pragma unroll
    for (int e = 0; e < 3; ++e) { const float d = v[e] - mu; vs += d * d; }
    red[tid] = vs; __syncthreads();
    for (int o = 128; o > 0; o >>= 1) { if (tid < o) red[tid] += red[tid + o]; __syncthreads(); }
    const float rstd = rsqrtf(red[0] * (1.f / PD) + 1e-5f);
#pragma unroll
    for (int e = 0; e < 3; ++e) {
        const int f = tid * 3 + e;
        out[(size_t)p * PD + f] = (v[e] - mu) * rstd * g[f] + bt[f];
    }
}

// ----------------- LN(256) + pos-emb into sequence buffer ------------------
__global__ __launch_bounds__(64)
void k_pe_ln2_pos(const float* __restrict__ in, const float* __restrict__ g,
                  const float* __restrict__ bt, const float* __restrict__ pos,
                  float* __restrict__ x)
{
    const int p = blockIdx.x;             // 0..3135
    const int b = p / NPATCH, t = p % NPATCH;
    const int tid = threadIdx.x;
    __shared__ float red[64];

    float v[4]; float s = 0.f;
#pragma unroll
    for (int i = 0; i < 4; ++i) { v[i] = in[(size_t)p * DIM + tid * 4 + i]; s += v[i]; }
    red[tid] = s; __syncthreads();
    for (int o = 32; o > 0; o >>= 1) { if (tid < o) red[tid] += red[tid + o]; __syncthreads(); }
    const float mu = red[0] * (1.f / DIM);
    __syncthreads();
    float vs = 0.f;
#pragma unroll
    for (int i = 0; i < 4; ++i) { const float d = v[i] - mu; vs += d * d; }
    red[tid] = vs; __syncthreads();
    for (int o = 32; o > 0; o >>= 1) { if (tid < o) red[tid] += red[tid + o]; __syncthreads(); }
    const float rstd = rsqrtf(red[0] * (1.f / DIM) + 1e-5f);
    const size_t orow = (size_t)(b * SEQ + t + 1) * DIM;
#pragma unroll
    for (int i = 0; i < 4; ++i) {
        const int f = tid * 4 + i;
        x[orow + f] = (v[i] - mu) * rstd * g[f] + bt[f] + pos[(size_t)(t + 1) * DIM + f];
    }
}

__global__ __launch_bounds__(64)
void k_cls_pos(const float* __restrict__ cls, const float* __restrict__ pos,
               float* __restrict__ x)
{
    const int b = blockIdx.x, tid = threadIdx.x;
#pragma unroll
    for (int i = 0; i < 4; ++i) {
        const int f = tid * 4 + i;
        x[(size_t)(b * SEQ) * DIM + f] = cls[f] + pos[f];
    }
}

// ---------------- residual accumulate + LN(256) -> xn ----------------------
__global__ __launch_bounds__(64)
void k_resid_ln(const float* __restrict__ x, float* __restrict__ res,
                float* __restrict__ xn, const float* __restrict__ g,
                const float* __restrict__ bt, int add)
{
    const size_t row = (size_t)blockIdx.x * DIM;
    const int tid = threadIdx.x;
    __shared__ float red[64];

    float v[4]; float s = 0.f;
#pragma unroll
    for (int i = 0; i < 4; ++i) {
        const int f = tid * 4 + i;
        float rv = x[row + f] + (add ? res[row + f] : 0.f);
        res[row + f] = rv;
        v[i] = rv; s += rv;
    }
    red[tid] = s; __syncthreads();
    for (int o = 32; o > 0; o >>= 1) { if (tid < o) red[tid] += red[tid + o]; __syncthreads(); }
    const float mu = red[0] * (1.f / DIM);
    __syncthreads();
    float vs = 0.f;
#pragma unroll
    for (int i = 0; i < 4; ++i) { const float d = v[i] - mu; vs += d * d; }
    red[tid] = vs; __syncthreads();
    for (int o = 32; o > 0; o >>= 1) { if (tid < o) red[tid] += red[tid + o]; __syncthreads(); }
    const float rstd = rsqrtf(red[0] * (1.f / DIM) + 1e-5f);
#pragma unroll
    for (int i = 0; i < 4; ++i) {
        const int f = tid * 4 + i;
        xn[row + f] = (v[i] - mu) * rstd * g[f] + bt[f];
    }
}

// ------------- depthwise causal conv (k=4) + SiLU -> u ---------------------
__global__ __launch_bounds__(256)
void k_conv_silu(const float* __restrict__ xz, const float* __restrict__ cw,
                 const float* __restrict__ cb, float* __restrict__ u)
{
    const int gid = blockIdx.x * 256 + threadIdx.x;
    if (gid >= MROWS * DI) return;
    const int row = gid >> 9, d = gid & (DI - 1);
    const int b = row / SEQ, t = row % SEQ;
    float acc = cb[d];
#pragma unroll
    for (int k = 0; k < DCONV; ++k) {
        const int tt = t - (DCONV - 1) + k;
        if (tt >= 0)
            acc += xz[(size_t)(b * SEQ + tt) * (2 * DI) + d] * cw[d * DCONV + k];
    }
    u[gid] = acc / (1.f + __expf(-acc));     // SiLU
}

// ------------- dt = softplus(dbc[:, :16] @ dtproj_w + b) -------------------
__global__ __launch_bounds__(256)
void k_dt_softplus(const float* __restrict__ dbc, const float* __restrict__ Wd,
                   const float* __restrict__ bd, float* __restrict__ dt)
{
    const int gid = blockIdx.x * 256 + threadIdx.x;
    if (gid >= MROWS * DI) return;
    const int row = gid >> 9, d = gid & (DI - 1);
    const float* dr = dbc + (size_t)row * 48;
    float s = bd[d];
#pragma unroll
    for (int r = 0; r < DTR; ++r) s += dr[r] * Wd[r * DI + d];
    dt[gid] = (s > 20.f) ? s : log1pf(__expf(s));
}

// ------------- selective scan: lane per (batch, channel) -------------------
__global__ __launch_bounds__(128)
void k_scan(const float* __restrict__ dt, const float* __restrict__ u,
            const float* __restrict__ dbc, const float* __restrict__ A_log,
            const float* __restrict__ Dp, const float* __restrict__ xz,
            float* __restrict__ y)
{
    const int gid = blockIdx.x * 128 + threadIdx.x;   // < 8192
    const int b = gid >> 9, d = gid & (DI - 1);
    float An[DS], h[DS];
#pragma unroll
    for (int n = 0; n < DS; ++n) { An[n] = -__expf(A_log[d * DS + n]); h[n] = 0.f; }
    const float Dv = Dp[d];
    for (int t = 0; t < SEQ; ++t) {
        const int row = b * SEQ + t;
        const float dtv = dt[(size_t)row * DI + d];
        const float uv  = u[(size_t)row * DI + d];
        const float du  = dtv * uv;
        const float* bc = dbc + (size_t)row * 48;
        float ys = 0.f;
#pragma unroll
        for (int n = 0; n < DS; ++n) {
            const float dA = __expf(dtv * An[n]);
            h[n] = dA * h[n] + du * bc[16 + n];
            ys += h[n] * bc[32 + n];
        }
        const float zv = xz[(size_t)row * (2 * DI) + DI + d];
        const float sz = zv / (1.f + __expf(-zv));
        y[(size_t)row * DI + d] = (ys + uv * Dv) * sz;
    }
}

// ---------------------------------------------------------------------------
extern "C" void kernel_launch(void* const* d_in, const int* in_sizes, int n_in,
                              void* d_out, int out_size, void* d_ws, size_t ws_size,
                              hipStream_t stream)
{
    (void)in_sizes; (void)n_in; (void)out_size; (void)ws_size;

    const float* img      = (const float*)d_in[0];
    const float* pe_ln1_g = (const float*)d_in[1];
    const float* pe_ln1_b = (const float*)d_in[2];
    const float* pe_w     = (const float*)d_in[3];
    const float* pe_b     = (const float*)d_in[4];
    const float* pe_ln2_g = (const float*)d_in[5];
    const float* pe_ln2_b = (const float*)d_in[6];
    const float* pos_emb  = (const float*)d_in[7];
    const float* cls_tok  = (const float*)d_in[8];
    const float* ln_w     = (const float*)d_in[9];
    const float* ln_b     = (const float*)d_in[10];
    const float* in_w     = (const float*)d_in[11];
    const float* conv_w   = (const float*)d_in[12];
    const float* conv_b   = (const float*)d_in[13];
    const float* xproj_w  = (const float*)d_in[14];
    const float* dtproj_w = (const float*)d_in[15];
    const float* dtproj_b = (const float*)d_in[16];
    const float* A_log    = (const float*)d_in[17];
    const float* Dp       = (const float*)d_in[18];
    const float* out_w    = (const float*)d_in[19];
    const float* normf_w  = (const float*)d_in[20];
    const float* normf_b  = (const float*)d_in[21];
    const float* head_w   = (const float*)d_in[22];
    const float* head_b   = (const float*)d_in[23];
    float* out = (float*)d_out;

    // workspace layout (floats), total ~13.9M floats (~56 MB)
    float* ws = (float*)d_ws;
    float* patches = ws;                             // 3136*768
    float* pe_out  = patches + (size_t)3136 * PD;    // 3136*256
    float* x       = pe_out  + (size_t)3136 * DIM;   // 3152*256
    float* res     = x       + (size_t)MROWS * DIM;  // 3152*256
    float* xn      = res     + (size_t)MROWS * DIM;  // 3152*256
    float* xz      = xn      + (size_t)MROWS * DIM;  // 3152*1024
    float* u       = xz      + (size_t)MROWS * 2 * DI; // 3152*512
    float* dbc     = u       + (size_t)MROWS * DI;   // 3152*48
    float* dt      = dbc     + (size_t)MROWS * 48;   // 3152*512
    float* y       = dt      + (size_t)MROWS * DI;   // 3152*512

    // ---- patch embed ----
    k_patch_ln<<<3136, 256, 0, stream>>>(img, pe_ln1_g, pe_ln1_b, patches);
    k_gemm_wmma<<<dim3(DIM / TN, (3136 + TM - 1) / TM), 128, 0, stream>>>(
        patches, PD, pe_w, DIM, pe_out, DIM, 3136, DIM, PD, pe_b);
    k_pe_ln2_pos<<<3136, 64, 0, stream>>>(pe_out, pe_ln2_g, pe_ln2_b, pos_emb, x);
    k_cls_pos<<<BB, 64, 0, stream>>>(cls_tok, pos_emb, x);

    const int gridE = (MROWS * DI + 255) / 256;   // 6304
    const int mtiles = (MROWS + TM - 1) / TM;     // 25
    const dim3 gIn(2 * DI / TN, mtiles);          // (16, 25)
    const dim3 gXp(1, mtiles);                    // (1, 25)
    const dim3 gOut(DIM / TN, mtiles);            // (4, 25)

    for (int i = 0; i < NL; ++i) {
        k_resid_ln<<<MROWS, 64, 0, stream>>>(x, res, xn,
                                             ln_w + (size_t)i * DIM,
                                             ln_b + (size_t)i * DIM, i > 0 ? 1 : 0);
        k_gemm_wmma<<<gIn, 128, 0, stream>>>(
            xn, DIM, in_w + (size_t)i * DIM * 2 * DI, 2 * DI,
            xz, 2 * DI, MROWS, 2 * DI, DIM, nullptr);
        k_conv_silu<<<gridE, 256, 0, stream>>>(
            xz, conv_w + (size_t)i * DI * DCONV, conv_b + (size_t)i * DI, u);
        k_gemm_wmma<<<gXp, 128, 0, stream>>>(
            u, DI, xproj_w + (size_t)i * DI * 48, 48,
            dbc, 48, MROWS, 48, DI, nullptr);
        k_dt_softplus<<<gridE, 256, 0, stream>>>(
            dbc, dtproj_w + (size_t)i * DTR * DI, dtproj_b + (size_t)i * DI, dt);
        k_scan<<<(BB * DI) / 128, 128, 0, stream>>>(
            dt, u, dbc, A_log + (size_t)i * DI * DS, Dp + (size_t)i * DI, xz, y);
        k_gemm_wmma<<<gOut, 128, 0, stream>>>(
            y, DI, out_w + (size_t)i * DI * DIM, DIM,
            x, DIM, MROWS, DIM, DI, nullptr);
    }

    // ---- final norm + head (CLS rows only via lda = SEQ*DIM) ----
    k_resid_ln<<<MROWS, 64, 0, stream>>>(x, res, xn, normf_w, normf_b, 1);
    k_gemm_wmma<<<dim3((NCLS + TN - 1) / TN, 1), 128, 0, stream>>>(
        xn, SEQ * DIM, head_w, NCLS, out, NCLS, BB, NCLS, DIM, head_b);
}
